// OrthogonalAttentionModule_55774445305900
// MI455X (gfx1250) — compile-verified, matrix operand order checked by reading, not run on hardware
//
#include <hip/hip_runtime.h>
#include <hip/hip_bf16.h>

// ---------------------------------------------------------------------------
// Problem constants (from reference): B=2, S=128, C=512, H=8, HD=64
// ---------------------------------------------------------------------------
static constexpr int Bc = 2;
static constexpr int S  = 128;
static constexpr int C  = 512;
static constexpr int H  = 8;
static constexpr int HD = 64;

typedef __attribute__((ext_vector_type(16))) _Float16 v16h;
typedef __attribute__((ext_vector_type(8)))  float    v8f;
typedef __attribute__((ext_vector_type(2)))  __fp16   v2hp;  // cvt_pkrtz result
typedef __attribute__((ext_vector_type(8)))  _Float16 v8h;

union VA { v16h v; _Float16 h[16]; uint4 u[2]; };
union VC { v8f  v; float    f[8];  };
union PK { v2hp p[8]; uint4 u[2]; };
union H8 { v8h  h; uint4 u; };

// reuse_a as a template constant (builtin flags are immediates); RA set on all
// but the last WMMA of a run sharing the same A operand (ISA OPSEL reuse hint).
template <bool RA>
__device__ __forceinline__ v8f wmma_f16_t(v16h a, v16h b, v8f c) {
  // D = A(16x32 f16) * B(32x16 f16) + C(16x16 f32)
  return __builtin_amdgcn_wmma_f32_16x16x32_f16(
      /*neg_a=*/false, a, /*neg_b=*/false, b,
      /*c_mod=*/(short)0, c, /*reuse_a=*/RA, /*reuse_b=*/false);
}

// Async global->LDS copy (CDNA5 GLOBAL_LOAD_ASYNC_TO_LDS_B128, ASYNCcnt).
// VDST = per-lane LDS byte address; VADDR = per-lane 64-bit global address.
// Generic pointers to LDS carry the LDS offset in addr[31:0] (ISA aperture map).
__device__ __forceinline__ void async_b128(void* lds, const void* g) {
  unsigned l = (unsigned)(uintptr_t)lds;
  asm volatile("global_load_async_to_lds_b128 %0, %1, off"
               :: "v"(l), "v"(g) : "memory");
}
__device__ __forceinline__ void async_wait0() {
  asm volatile("s_wait_asynccnt 0x0" ::: "memory");
}

// Inverse of the documented 16-bit 16x32 fragment layout:
//   lane = m/n_local + (k&8 ? 16 : 0)
//   elem = 2*((k>>4)*4 + ((k&7)>>1)) + (k&1)
// NB: this collapses to four contiguous 8-half groups per 32-K row run:
//   K0-7 -> lane ml  elems 0-7 ; K8-15 -> lane ml+16 elems 0-7
//   K16-23 -> lane ml elems 8-15 ; K24-31 -> lane ml+16 elems 8-15
__device__ __forceinline__ int frag_elem(int kl) {
  return 2 * ((((kl >> 4) & 1) << 2) + ((kl & 7) >> 1)) + (kl & 1);
}

// Fragment stored as [lane][16 halfs] (512 halfs / fragment): contiguous
// 32B per lane -> two ds_load_b128 per fragment, no packing movs.
__device__ __forceinline__ v16h load_frag(const _Float16* fb, int lane) {
  VA u;
  const uint4* p = (const uint4*)(fb + (size_t)lane * 16);
  u.u[0] = p[0];
  u.u[1] = p[1];
  return u.v;
}

// ---------------------------------------------------------------------------
// Kernel 0: weight conversion straight into *global fragment order*.
//   WqF:  [kb(16)][ntile(96)][lane(32)][elem(16)]   (B-frags of W_qkv)
//   WtF:  [kb(16)][ntile(32)][lane(32)][elem(16)]   (B-frags of W_out^T)
// ---------------------------------------------------------------------------
__global__ __launch_bounds__(256) void k_convert(const float* __restrict__ Wqkv,
                                                 const float* __restrict__ Wout,
                                                 _Float16* __restrict__ WqF,
                                                 _Float16* __restrict__ WtF) {
  int idx = blockIdx.x * 256 + threadIdx.x;
  const int NQ = C * 3 * C;  // 786432
  if (idx < NQ) {
    int c = idx / (3 * C), n = idx % (3 * C);
    int kb = c >> 5, kl = c & 31, nt = n >> 4, nl = n & 15;
    WqF[((size_t)kb * 96 + nt) * 512 +
        (nl + ((kl & 8) ? 16 : 0)) * 16 + frag_elem(kl)] = (_Float16)Wqkv[idx];
  } else {
    int j = idx - NQ;  // 0 .. 262143
    int c = j >> 9, n = j & 511;  // B[k=c][n] = W_out[n][c]
    int kb = c >> 5, kl = c & 31, nt = n >> 4, nl = n & 15;
    WtF[((size_t)kb * 32 + nt) * 512 +
        (nl + ((kl & 8) ? 16 : 0)) * 16 + frag_elem(kl)] =
        (_Float16)Wout[(size_t)n * C + c];
  }
}

// ---------------------------------------------------------------------------
// Kernel 1: QKV GEMM.  X (98304 x 512) fp32 * Wq (512 x 1536) -> qkv fp16
// qkv layout: [t(3)][b][plane(3)][h][i][j][d]  (d contiguous, 64 halfs)
// grid (768, 12): blockIdx.x = (b*3+p)*S + i ; blockIdx.y = n-tile of 128
// ---------------------------------------------------------------------------
__global__ __launch_bounds__(256) void k_qkv(const float* __restrict__ pxy,
                                             const float* __restrict__ pxz,
                                             const float* __restrict__ pyz,
                                             const _Float16* __restrict__ WqF,
                                             _Float16* __restrict__ qkv) {
  __shared__ __align__(16) _Float16 AsF[8 * 512];  // 8 m-frags
  __shared__ __align__(16) _Float16 BsF[8 * 512];  // 8 n-frags

  int mt = blockIdx.x;
  int i  = mt & (S - 1);
  int bp = mt >> 7;  // b*3 + p
  int p  = bp % 3;
  int b  = bp / 3;
  const float* Xp = (p == 0) ? pxy : ((p == 1) ? pxz : pyz);
  const float* X  = Xp + ((size_t)(b * S + i)) * S * C;  // 128 rows (j)

  int tid = threadIdx.x, wave = tid >> 5, lane = tid & 31;
  int n0 = blockIdx.y * 128;
  size_t nt0 = (size_t)(n0 >> 4);  // first of 8 B n-frags

  int row = tid >> 1, ml = row & 15;
  int hsel = tid & 1;  // selects K 0..15 (0) or 16..31 (1) of the 32-K step
  _Float16* afb = AsF + (row >> 4) * 512;
  _Float16* d0 = afb + ml * 16 + hsel * 8;         // K-low 8 of this half
  _Float16* d1 = afb + (ml + 16) * 16 + hsel * 8;  // K-high 8 of this half

  VC acc[8];
#pragma unroll
  for (int ct = 0; ct < 8; ++ct)
#pragma unroll
    for (int r = 0; r < 8; ++r) acc[ct].f[r] = 0.f;

  for (int k0 = 0; k0 < C; k0 += 32) {
    {  // stage B first: async DMA, fragment-ordered weights (no VGPR hop)
      const _Float16* sp = WqF + ((size_t)(k0 >> 5) * 96 + nt0) * 512 +
                           (size_t)tid * 16;
      async_b128(BsF + (size_t)tid * 16, sp);
      async_b128(BsF + (size_t)tid * 16 + 8, sp + 8);
    }
    {  // stage A: f32 -> packed f16 (cvt_pk) -> two ds_store_b128
      const float* src = X + (size_t)row * C + k0 + hsel * 16;
      __builtin_prefetch((const void*)(src + 32), 0, 3);
      float4 f0 = *(const float4*)(src + 0);
      float4 f1 = *(const float4*)(src + 4);
      float4 f2 = *(const float4*)(src + 8);
      float4 f3 = *(const float4*)(src + 12);
      PK pk;
      pk.p[0] = __builtin_amdgcn_cvt_pkrtz(f0.x, f0.y);
      pk.p[1] = __builtin_amdgcn_cvt_pkrtz(f0.z, f0.w);
      pk.p[2] = __builtin_amdgcn_cvt_pkrtz(f1.x, f1.y);
      pk.p[3] = __builtin_amdgcn_cvt_pkrtz(f1.z, f1.w);
      pk.p[4] = __builtin_amdgcn_cvt_pkrtz(f2.x, f2.y);
      pk.p[5] = __builtin_amdgcn_cvt_pkrtz(f2.z, f2.w);
      pk.p[6] = __builtin_amdgcn_cvt_pkrtz(f3.x, f3.y);
      pk.p[7] = __builtin_amdgcn_cvt_pkrtz(f3.z, f3.w);
      *(uint4*)d0 = pk.u[0];
      *(uint4*)d1 = pk.u[1];
    }
    async_wait0();
    __syncthreads();
    v16h a = load_frag(AsF + wave * 512, lane);
#pragma unroll
    for (int ct = 0; ct < 8; ++ct) {
      v16h bb = load_frag(BsF + ct * 512, lane);
      if (ct < 7) acc[ct].v = wmma_f16_t<true>(a, bb, acc[ct].v);
      else        acc[ct].v = wmma_f16_t<false>(a, bb, acc[ct].v);
    }
    __syncthreads();
  }

  const size_t PB = (size_t)S * S * HD;  // per (t,b,p,h) block
#pragma unroll
  for (int ct = 0; ct < 8; ++ct) {
    int n = n0 + ct * 16 + (lane & 15);
    int t = n >> 9, h = (n >> 6) & 7, d = n & 63;
    size_t base = ((((size_t)t * Bc + b) * 3 + p) * H + h) * PB +
                  (size_t)i * S * HD + d;
#pragma unroll
    for (int r = 0; r < 8; ++r) {
      int j = wave * 16 + r + ((lane >> 4) << 3);
      qkv[base + (size_t)j * HD] = (_Float16)acc[ct].f[r];
    }
  }
}

// ---------------------------------------------------------------------------
// Kernel 2: six attention types. One block = one (type, b, h, shared index s).
// All LDS tiles are in fragment order.
//   QF: [mtile(8)][kb(2)][512]   KF: [ntile(8)][kb(2)][512]
//   VF: [ntile(4)][kb(4)][512]   PF: [mtile(8)][kb(4)][512] (over QF+KF)
// ---------------------------------------------------------------------------
__global__ __launch_bounds__(256) void k_attn(const _Float16* __restrict__ qkv,
                                              _Float16* __restrict__ terms) {
  __shared__ __align__(16) _Float16 smem[3 * 8192];  // 48 KB
  _Float16* VF = smem;          // 8192 halfs
  _Float16* QF = smem + 8192;   // 8192
  _Float16* KF = smem + 16384;  // 8192
  _Float16* PF = smem + 8192;   // 16384, reuses Q/K region after barrier

  int bid = blockIdx.x;
  int s = bid & 127;
  int h = (bid >> 7) & 7;
  int b = (bid >> 10) & 1;
  int type = bid >> 11;  // 0..5

  const int qp_[6] = {0, 0, 1, 1, 2, 2};
  const int kp_[6] = {1, 2, 0, 2, 0, 1};
  const int qs_[6] = {0, 1, 0, 1, 0, 1};  // queries along i (strided)?
  const int ks_[6] = {0, 0, 0, 1, 1, 1};  // keys along i (strided)?
  int qp = qp_[type], kp = kp_[type], qstr = qs_[type], kstr = ks_[type];

  const size_t PB = (size_t)S * S * HD;
  const _Float16* qb = qkv + ((((size_t)0 * Bc + b) * 3 + qp) * H + h) * PB;
  const _Float16* kb_ = qkv + ((((size_t)1 * Bc + b) * 3 + kp) * H + h) * PB;
  const _Float16* vb = qkv + ((((size_t)2 * Bc + b) * 3 + kp) * H + h) * PB;
  size_t qoff = qstr ? (size_t)s * HD : (size_t)s * S * HD;
  size_t koff = kstr ? (size_t)s * HD : (size_t)s * S * HD;
  int qst = qstr ? S * HD : HD;
  int kst = kstr ? S * HD : HD;

  int tid = threadIdx.x, wave = tid >> 5, lane = tid & 31;

  {  // stage Q, K, V into fragment-ordered LDS
    int row  = tid >> 1;       // 0..127
    int hsel = tid & 1;        // half of the 64 d-columns (also kb index)
    int cb   = hsel * 32;
    int rl   = row & 15;

    // Q: four contiguous b128 groups per 32-K run -> async DMA
    const _Float16* gq = qb + qoff + (size_t)row * qst + cb;
    _Float16* fq = QF + (row >> 4) * 1024 + hsel * 512;
    async_b128(fq + rl * 16,            gq);       // K 0-7
    async_b128(fq + (rl + 16) * 16,     gq + 8);   // K 8-15
    async_b128(fq + rl * 16 + 8,        gq + 16);  // K 16-23
    async_b128(fq + (rl + 16) * 16 + 8, gq + 24);  // K 24-31

    // K used transposed (B[k=d][n=key]) -> identical scatter formula
    const _Float16* gk = kb_ + koff + (size_t)row * kst + cb;
    _Float16* fk = KF + (row >> 4) * 1024 + hsel * 512;
    async_b128(fk + rl * 16,            gk);
    async_b128(fk + (rl + 16) * 16,     gk + 8);
    async_b128(fk + rl * 16 + 8,        gk + 16);
    async_b128(fk + (rl + 16) * 16 + 8, gk + 24);

    // V as B[k=key][n=d]: transpose direction -> scalar scatter (hoisted)
    union { uint4 u[4]; _Float16 h[32]; } t;
    const uint4* sv = (const uint4*)(vb + koff + (size_t)row * kst + cb);
#pragma unroll
    for (int q = 0; q < 4; ++q) t.u[q] = sv[q];
    int kb2 = row >> 5;
    int le  = (row & 8) ? 16 : 0;
    int er  = frag_elem(row & 31);
    _Float16* fv = VF + kb2 * 512 + er;
#pragma unroll
    for (int q = 0; q < 32; ++q)
      fv[(((cb + q) >> 4) * 2048) + ((q & 15) + le) * 16] = t.h[q];
  }
  async_wait0();
  __syncthreads();

  // ---- scores = Q * K^T ----
  VC acc[8];
#pragma unroll
  for (int ct = 0; ct < 8; ++ct)
#pragma unroll
    for (int r = 0; r < 8; ++r) acc[ct].f[r] = 0.f;
#pragma unroll
  for (int kb = 0; kb < 2; ++kb) {
    v16h a = load_frag(QF + wave * 1024 + kb * 512, lane);
#pragma unroll
    for (int ct = 0; ct < 8; ++ct) {
      v16h bb = load_frag(KF + ct * 1024 + kb * 512, lane);
      if (ct < 7) acc[ct].v = wmma_f16_t<true>(a, bb, acc[ct].v);
      else        acc[ct].v = wmma_f16_t<false>(a, bb, acc[ct].v);
    }
  }

  // ---- softmax over 128 key columns (rows split across 16-lane halves) ----
  const float scale = 0.125f;  // hd^-0.5
#pragma unroll
  for (int r = 0; r < 8; ++r) {
    float mx = acc[0].f[r];
#pragma unroll
    for (int ct = 1; ct < 8; ++ct) mx = fmaxf(mx, acc[ct].f[r]);
    mx = fmaxf(mx, __shfl_xor(mx, 8, 32));
    mx = fmaxf(mx, __shfl_xor(mx, 4, 32));
    mx = fmaxf(mx, __shfl_xor(mx, 2, 32));
    mx = fmaxf(mx, __shfl_xor(mx, 1, 32));
    float sum = 0.f;
#pragma unroll
    for (int ct = 0; ct < 8; ++ct) {
      float e = __expf((acc[ct].f[r] - mx) * scale);
      acc[ct].f[r] = e;
      sum += e;
    }
    sum += __shfl_xor(sum, 8, 32);
    sum += __shfl_xor(sum, 4, 32);
    sum += __shfl_xor(sum, 2, 32);
    sum += __shfl_xor(sum, 1, 32);
    float inv = 1.0f / sum;
#pragma unroll
    for (int ct = 0; ct < 8; ++ct) acc[ct].f[r] *= inv;
  }

  __syncthreads();  // all waves done reading Q/K before P overwrites them

  // ---- write probabilities into fragment-ordered PF (transpose: scalar) ----
  {
    int lhi = (lane >> 4) << 3;  // 0 or 8
    int ln  = lane & 15;
    _Float16* pw = PF + wave * 2048;
#pragma unroll
    for (int ct = 0; ct < 8; ++ct) {
      int kb = ct >> 1;
      int kl = ((ct & 1) << 4) + ln;
      int eo = frag_elem(kl);
      int lo = (ln & 8) ? 16 : 0;
#pragma unroll
      for (int r = 0; r < 8; ++r)
        pw[kb * 512 + ((r + lhi) + lo) * 16 + eo] = (_Float16)acc[ct].f[r];
    }
  }
  __syncthreads();

  // ---- O = P * V ----
  VC o[4];
#pragma unroll
  for (int ct = 0; ct < 4; ++ct)
#pragma unroll
    for (int r = 0; r < 8; ++r) o[ct].f[r] = 0.f;
#pragma unroll
  for (int kb = 0; kb < 4; ++kb) {
    v16h a = load_frag(PF + wave * 2048 + kb * 512, lane);
#pragma unroll
    for (int ct = 0; ct < 4; ++ct) {
      v16h bb = load_frag(VF + ct * 2048 + kb * 512, lane);
      if (ct < 3) o[ct].v = wmma_f16_t<true>(a, bb, o[ct].v);
      else        o[ct].v = wmma_f16_t<false>(a, bb, o[ct].v);
    }
  }

  // ---- store into term buffer with ca/cb mapping ----
  _Float16* T = terms + (size_t)type * ((size_t)Bc * S * S * C);
  int m0 = wave * 16;
#pragma unroll
  for (int ct = 0; ct < 4; ++ct) {
    int d = ct * 16 + (lane & 15);
#pragma unroll
    for (int r = 0; r < 8; ++r) {
      int q = m0 + r + ((lane >> 4) << 3);
      int ii = qstr ? q : s;
      int jj = qstr ? s : q;
      size_t idx = (((size_t)b * S + ii) * S + jj) * C + h * HD + d;
      T[idx] = (_Float16)o[ct].f[r];
    }
  }
}

// ---------------------------------------------------------------------------
// Kernel 3: output projection. Y = (termA+termB) @ W_out^T + b_out, fp32 out.
// blockIdx.x = ((plane*256 + mtile)*4 + ntile)
// ---------------------------------------------------------------------------
__global__ __launch_bounds__(256) void k_proj(const _Float16* __restrict__ terms,
                                              const _Float16* __restrict__ WtF,
                                              const float* __restrict__ bias,
                                              float* __restrict__ out) {
  __shared__ __align__(16) _Float16 AsF[8 * 512];
  __shared__ __align__(16) _Float16 BsF[8 * 512];

  int id = blockIdx.x;
  int nt = id & 3;
  int t2 = id >> 2;
  int mt = t2 & 255;
  int plane = t2 >> 8;

  const size_t PLANE = (size_t)Bc * S * S * C;
  const _Float16* TA = terms + (size_t)(2 * plane) * PLANE;
  const _Float16* TB = terms + (size_t)(2 * plane + 1) * PLANE;
  size_t m0 = (size_t)mt * 128;
  int n0 = nt * 128;
  size_t nt0 = (size_t)(n0 >> 4);

  int tid = threadIdx.x, wave = tid >> 5, lane = tid & 31;
  int row = tid >> 1, ml = row & 15;
  int hsel = tid & 1;
  _Float16* afb = AsF + (row >> 4) * 512;
  _Float16* d0 = afb + ml * 16 + hsel * 8;
  _Float16* d1 = afb + (ml + 16) * 16 + hsel * 8;

  VC acc[8];
#pragma unroll
  for (int ct = 0; ct < 8; ++ct)
#pragma unroll
    for (int r = 0; r < 8; ++r) acc[ct].f[r] = 0.f;

  for (int k0 = 0; k0 < C; k0 += 32) {
    {  // stage B: async DMA of fragment-ordered W_out^T
      const _Float16* sp = WtF + ((size_t)(k0 >> 5) * 32 + nt0) * 512 +
                           (size_t)tid * 16;
      async_b128(BsF + (size_t)tid * 16, sp);
      async_b128(BsF + (size_t)tid * 16 + 8, sp + 8);
    }
    {  // stage A = termA + termB with packed half adds, two b128 stores
      const _Float16* a0 = TA + (m0 + row) * C + k0 + hsel * 16;
      const _Float16* a1 = TB + (m0 + row) * C + k0 + hsel * 16;
      __builtin_prefetch((const void*)(a0 + 32), 0, 3);
      __builtin_prefetch((const void*)(a1 + 32), 0, 3);
      H8 x0, x1, y0, y1;
      x0.u = *(const uint4*)(a0);
      x1.u = *(const uint4*)(a0 + 8);
      y0.u = *(const uint4*)(a1);
      y1.u = *(const uint4*)(a1 + 8);
      x0.h += y0.h;  // v_pk_add_f16
      x1.h += y1.h;
      *(uint4*)d0 = x0.u;
      *(uint4*)d1 = x1.u;
    }
    async_wait0();
    __syncthreads();
    v16h a = load_frag(AsF + wave * 512, lane);
#pragma unroll
    for (int ct = 0; ct < 8; ++ct) {
      v16h bb = load_frag(BsF + ct * 512, lane);
      if (ct < 7) acc[ct].v = wmma_f16_t<true>(a, bb, acc[ct].v);
      else        acc[ct].v = wmma_f16_t<false>(a, bb, acc[ct].v);
    }
    __syncthreads();
  }

  float* O = out + (size_t)plane * PLANE;
#pragma unroll
  for (int ct = 0; ct < 8; ++ct) {
    int n = n0 + ct * 16 + (lane & 15);
    float bv = bias[n];
#pragma unroll
    for (int r = 0; r < 8; ++r) {
      int r2 = wave * 16 + r + ((lane >> 4) << 3);
      O[(m0 + r2) * C + n] = acc[ct].f[r] + bv;
    }
  }
}

// ---------------------------------------------------------------------------
// Host-side launch. Workspace layout (bytes):
//   [0)      qkv fp16:   3*2*3*8*128*128*64 halfs = 301,989,888 B
//   [qkv)    terms fp16: 6*2*128*128*512 halfs    = 201,326,592 B
//   [terms)  WqF fp16:   786,432 halfs            =   1,572,864 B
//   [WqF)    WtF fp16:   262,144 halfs            =     524,288 B
// ---------------------------------------------------------------------------
extern "C" void kernel_launch(void* const* d_in, const int* in_sizes, int n_in,
                              void* d_out, int out_size, void* d_ws, size_t ws_size,
                              hipStream_t stream) {
  const float* pxy  = (const float*)d_in[0];
  const float* pxz  = (const float*)d_in[1];
  const float* pyz  = (const float*)d_in[2];
  const float* Wqkv = (const float*)d_in[3];
  const float* Wout = (const float*)d_in[4];
  const float* bout = (const float*)d_in[5];
  float* outp = (float*)d_out;

  static constexpr size_t QKV_BYTES  = (size_t)3 * Bc * 3 * H * S * S * HD * 2;
  static constexpr size_t TERM_BYTES = (size_t)6 * Bc * S * S * C * 2;
  static constexpr size_t WQ_BYTES   = (size_t)C * 3 * C * 2;

  char* ws = (char*)d_ws;
  _Float16* qkv   = (_Float16*)ws;
  _Float16* terms = (_Float16*)(ws + QKV_BYTES);
  _Float16* WqF   = (_Float16*)(ws + QKV_BYTES + TERM_BYTES);
  _Float16* WtF   = (_Float16*)(ws + QKV_BYTES + TERM_BYTES + WQ_BYTES);

  // 0) weight conversion into fragment order: exactly 1,048,576 elements
  k_convert<<<4096, 256, 0, stream>>>(Wqkv, Wout, WqF, WtF);

  // 1) QKV GEMM: 768 m-tiles (one per (b,plane,i)) x 12 n-tiles
  k_qkv<<<dim3(768, 12), 256, 0, stream>>>(pxy, pxz, pyz, WqF, qkv);

  // 2) attention: 6 types * B * H * S = 12288 instances
  k_attn<<<12288, 256, 0, stream>>>(qkv, terms);

  // 3) projection: 3 planes * 256 m-tiles * 4 n-tiles
  k_proj<<<3072, 256, 0, stream>>>(terms, WtF, bout, outp);
}
